// MambaFusedWrapper_27281632264304
// MI455X (gfx1250) — compile-verified, matrix operand order checked by reading, not run on hardware
//
#include <hip/hip_runtime.h>
#include <hip/hip_bf16.h>

// ---------------- problem constants (from setup_inputs) ----------------
static constexpr int L_SEQ  = 2048;
static constexpr int NHEAD  = 32;
static constexpr int HDIM   = 64;     // P
static constexpr int DSTATE = 128;    // N
static constexpr int DIN    = NHEAD * HDIM;          // 2048
static constexpr int CCH    = DIN + 2 * DSTATE;      // 2304 conv channels
static constexpr int DPROJ  = 2 * DIN + 2 * DSTATE + NHEAD; // 4384
static constexpr int TCH    = 256;    // chunk size
static constexpr int NCHUNK = L_SEQ / TCH;
static constexpr int KCONV  = 4;

typedef __attribute__((ext_vector_type(16))) __bf16 v16bf;
typedef __attribute__((ext_vector_type(8)))  float  v8f;
typedef __attribute__((ext_vector_type(4)))  unsigned v4u;
typedef __attribute__((ext_vector_type(8)))  int    v8i;
typedef __attribute__((ext_vector_type(4)))  int    v4i;

// ---------------- small helpers ----------------
__device__ __forceinline__ unsigned short f2bfbits(float f) {
  unsigned u = __float_as_uint(f);
  unsigned r = u + 0x7FFFu + ((u >> 16) & 1u);   // round-to-nearest-even
  return (unsigned short)(r >> 16);
}
__device__ __forceinline__ float bfbits2f(unsigned short h) {
  return __uint_as_float(((unsigned)h) << 16);
}
__device__ __forceinline__ __bf16 bits2bf(unsigned short h) {
  __bf16 b; __builtin_memcpy(&b, &h, 2); return b;
}
// K index held by element e of an A/B fragment (wave32, 16-bit 16x32 layout)
__device__ __forceinline__ int kmap(int e, int hi) {
  return ((e < 8) ? e : e + 8) + (hi ? 8 : 0);
}
__device__ __forceinline__ v8f vzero8() {
  v8f v;
#pragma unroll
  for (int i = 0; i < 8; i++) v[i] = 0.f;
  return v;
}
__device__ __forceinline__ v8f wmma_bf16(v16bf a, v16bf b, v8f c) {
  return __builtin_amdgcn_wmma_f32_16x16x32_bf16(false, a, false, b, (short)0, c,
                                                 false, false);
}
// fragment whose 16 K-elements are CONTIGUOUS in memory (k = fastest dim)
__device__ __forceinline__ v16bf frag_krow(const unsigned short* p, int hi) {
  v16bf a;
#pragma unroll
  for (int j = 0; j < 8; j++) {
    int k0 = kmap(2 * j, hi);                      // even; k0+1 = kmap(2j+1)
    unsigned v = *(const unsigned*)(p + k0);       // 4B-aligned dword pair
    a[2 * j]     = bits2bf((unsigned short)(v & 0xffffu));
    a[2 * j + 1] = bits2bf((unsigned short)(v >> 16));
  }
  return a;
}
// fragment whose K-elements are strided (k indexes rows)
__device__ __forceinline__ v16bf frag_kstride(const unsigned short* p,
                                              int stride, int hi) {
  v16bf a;
#pragma unroll
  for (int e = 0; e < 16; e++) a[e] = bits2bf(p[(size_t)kmap(e, hi) * stride]);
  return a;
}

// ---- TDM: 2-D tile (rows x dim0 of 2-byte elements) global -> LDS ----
// Builds D# group0/group1 per CDNA5 ISA 8.3/8.4; groups 2/3 zero (2-D tensor).
__device__ __forceinline__ void tdm_load_2d_b16(unsigned lds_off, unsigned long long ga,
                                                unsigned dim0, unsigned rows,
                                                unsigned stride0) {
  v4u g0;
  g0[0] = 1u;                                        // count=1 (valid), user mode
  g0[1] = lds_off;                                   // lds_addr (bytes)
  g0[2] = (unsigned)(ga & 0xffffffffu);              // global_addr[31:0]
  g0[3] = (unsigned)((ga >> 32) & 0x1ffffffu) | (2u << 30);  // addr[56:32] | type=2
  v8i g1;
  g1[0] = (int)(1u << 16);                           // data_size=1 -> 2 bytes
  g1[1] = (int)(dim0 << 16);                         // tensor_dim0[15:0] @bits63:48
  g1[2] = (int)(((rows & 0xffffu) << 16) | (dim0 >> 16)); // dim1 lo | dim0 hi
  g1[3] = (int)((dim0 << 16) | (rows >> 16));        // tile_dim0 | dim1 hi
  g1[4] = (int)(rows & 0xffffu);                     // tile_dim1 (tile_dim2=0)
  g1[5] = (int)stride0;                              // tensor_dim0_stride[31:0]
  g1[6] = 0;                                         // stride0 hi | stride1 lo
  g1[7] = 0;
  v4i gz; gz[0] = 0; gz[1] = 0; gz[2] = 0; gz[3] = 0;
  v8i gz8;
#pragma unroll
  for (int i = 0; i < 8; i++) gz8[i] = 0;
  __builtin_amdgcn_tensor_load_to_lds(g0, g1, gz, gz, gz8, 0);
}

// ---------------- kernel 1: depthwise causal conv + SiLU + split ----------------
// x is written BOTH as fp32 [b,l,h*p] (for the D*x epilogue) and as bf16
// TRANSPOSED [b,h,p,t] so the SSD kernel's X tiles have contiguous K=t.
__global__ void conv_silu_kernel(const float* __restrict__ zx,
                                 const float* __restrict__ cw,
                                 const float* __restrict__ cb,
                                 float* __restrict__ xf,
                                 unsigned short* __restrict__ xbfT,
                                 unsigned short* __restrict__ Bbf,
                                 unsigned short* __restrict__ Cbf,
                                 int total) {
  int idx = blockIdx.x * blockDim.x + threadIdx.x;
  if (idx >= total) return;
  int cch = idx % CCH;
  int bl  = idx / CCH;                 // b*L + l
  int l   = bl % L_SEQ;
  int b   = bl / L_SEQ;
  float acc = cb[cch];
  const float* base = zx + (size_t)b * L_SEQ * DPROJ + DIN + cch;
#pragma unroll
  for (int k = 0; k < KCONV; k++) {
    int lt = l - (KCONV - 1) + k;
    if (lt >= 0) acc += base[(size_t)lt * DPROJ] * cw[cch * KCONV + k];
  }
  float s = acc / (1.f + __expf(-acc));   // SiLU
  if (cch < DIN) {
    xf[(size_t)bl * DIN + cch] = s;
    int h = cch / HDIM, p = cch % HDIM;
    xbfT[((size_t)(b * NHEAD + h) * HDIM + p) * L_SEQ + l] = f2bfbits(s);
  } else if (cch < DIN + DSTATE) {
    Bbf[(size_t)bl * DSTATE + (cch - DIN)] = f2bfbits(s);
  } else {
    Cbf[(size_t)bl * DSTATE + (cch - DIN - DSTATE)] = f2bfbits(s);
  }
}

// ---------------- kernel 2: dt = clip(softplus(...)), logdA = dt * A ----------------
__global__ void dt_kernel(const float* __restrict__ zx,
                          const float* __restrict__ dt_bias,
                          const float* __restrict__ a_log,
                          const float* __restrict__ dt_scale,
                          const int* __restrict__ dtmin_p,
                          const int* __restrict__ dtmax_p,
                          float* __restrict__ dtw,
                          float* __restrict__ dAl,
                          int total) {
  int idx = blockIdx.x * blockDim.x + threadIdx.x;
  if (idx >= total) return;
  int h  = idx % NHEAD;
  int bl = idx / NHEAD;
  float raw = zx[(size_t)bl * DPROJ + (DPROJ - NHEAD) + h];
  float v = raw * dt_scale[h] + dt_bias[h];
  float sp = (v > 20.f) ? v : log1pf(__expf(v));
  sp = fminf(fmaxf(sp, (float)dtmin_p[0]), (float)dtmax_p[0]);
  dtw[idx] = sp;
  dAl[idx] = -sp * __expf(a_log[h]);
}

// ---------------- kernel 3: chunked SSD scan, TDM-staged LDS + WMMA ----------------
__global__ void __launch_bounds__(256)
ssd_chunk_kernel(const float* __restrict__ zx,
                 const float* __restrict__ xf,
                 const unsigned short* __restrict__ xbfT,
                 const unsigned short* __restrict__ Bbf,
                 const unsigned short* __restrict__ Cbf,
                 const float* __restrict__ dtw,
                 const float* __restrict__ dAl,
                 const float* __restrict__ d_param,
                 const float* __restrict__ init_state,
                 float* __restrict__ out) {
  const int bh = blockIdx.x;
  const int b  = bh / NHEAD;
  const int h  = bh % NHEAD;
  const int tid    = threadIdx.x;
  const int wave   = tid >> 5;
  const int lane   = tid & 31;
  const int lane16 = lane & 15;
  const int laneHi = (lane >> 4) & 1;

  __shared__ float sS[HDIM * DSTATE];              // 64x128 running state (32 KB)
  __shared__ unsigned short sB[TCH * DSTATE];      // chunk B tile [t][n] (64 KB)
  __shared__ unsigned short sXT[HDIM * TCH];       // chunk X^T tile [p][t] (32 KB)
  __shared__ float sCum[TCH], sDt[TCH], sW[TCH];   // (3 KB)
  __shared__ unsigned short sM[8][16 * 32];        // per-wave score panel (8 KB)

  // init state from initial_state (B,H,P,N)
  {
    const float* s0 = init_state + (size_t)bh * HDIM * DSTATE;
    for (int i = tid; i < HDIM * DSTATE; i += 256) sS[i] = s0[i];
  }
  __syncthreads();

  const float dpar = d_param[h];
  const unsigned ldsB = (unsigned)(size_t)(&sB[0]);
  const unsigned ldsX = (unsigned)(size_t)(&sXT[0]);

  for (int c = 0; c < NCHUNK; c++) {
    const int l0 = c * TCH;
    const unsigned short* Crow0 = Cbf + (size_t)(b * L_SEQ + l0) * DSTATE;

    // ---- kick off TDM DMA: B tile (256x128, [t][n]) and X^T tile (64x256, [p][t])
    if (wave == 0) {
      unsigned long long gB =
          (unsigned long long)(size_t)(Bbf + (size_t)(b * L_SEQ + l0) * DSTATE);
      unsigned long long gX =
          (unsigned long long)(size_t)(xbfT +
                                       (size_t)(b * NHEAD + h) * HDIM * L_SEQ + l0);
      tdm_load_2d_b16(ldsB, gB, DSTATE, TCH, DSTATE);  // rows=t, contiguous n
      tdm_load_2d_b16(ldsX, gX, TCH, HDIM, L_SEQ);     // rows=p, contiguous t
    }

    // ---- phase 0 (overlaps TDM): decay prefix sums + weights ----
    sDt[tid]  = dtw[(size_t)(b * L_SEQ + l0 + tid) * NHEAD + h];
    sCum[tid] = dAl[(size_t)(b * L_SEQ + l0 + tid) * NHEAD + h];
    __syncthreads();
    if (tid == 0) {
      float run = 0.f;
      for (int t = 0; t < TCH; t++) { run += sCum[t]; sCum[t] = run; }
    }
    __syncthreads();
    const float cumTot = sCum[TCH - 1];
    sW[tid] = sDt[tid] * __expf(cumTot - sCum[tid]);   // state-update weight

    if (wave == 0) __builtin_amdgcn_s_wait_tensorcnt(0);  // issuing wave drains TDM
    __syncthreads();                                      // publish LDS tiles + sW

    // ---- phase A: outputs (reads sS/sB/sXT, never writes them) ----
    for (int band = wave; band < TCH / 16; band += 8) {
      const int mrowBase = band * 16;
      // A operand: C rows of this band, K = n (0..127), from global (dword pairs)
      v16bf aC[4];
#pragma unroll
      for (int kk = 0; kk < 4; kk++)
        aC[kk] = frag_krow(Crow0 + (size_t)(mrowBase + lane16) * DSTATE + kk * 32,
                           laneHi);

      v8f acc[4];
#pragma unroll
      for (int nt = 0; nt < 4; nt++) acc[nt] = vzero8();

      // Y_inter = C * S_prev^T   (K = n)
#pragma unroll
      for (int kk = 0; kk < 4; kk++) {
#pragma unroll
        for (int nt = 0; nt < 4; nt++) {
          v16bf bS;
#pragma unroll
          for (int e = 0; e < 16; e++) {
            int k = kk * 32 + kmap(e, laneHi);
            bS[e] = bits2bf(f2bfbits(sS[(nt * 16 + lane16) * DSTATE + k]));
          }
          acc[nt] = wmma_bf16(aC[kk], bS, acc[nt]);
        }
      }
      // scale Y_inter rows by exp(cum[t])
#pragma unroll
      for (int nt = 0; nt < 4; nt++)
#pragma unroll
        for (int r = 0; r < 8; r++) {
          int mrow = r + (laneHi ? 8 : 0);
          acc[nt][r] *= __expf(sCum[mrowBase + mrow]);
        }

      // Y_intra: stream K=32 panels of M = mask∘decay∘(C B^T) into Y += M X
      unsigned short* sMw = sM[wave];
      for (int jj = 0; jj <= band; jj += 2) {
        const bool two = (jj + 1 <= band);
        v8f g0 = vzero8(), g1 = vzero8();
#pragma unroll
        for (int kk = 0; kk < 4; kk++) {
          v16bf bB0 = frag_krow(&sB[(jj * 16 + lane16) * DSTATE + kk * 32], laneHi);
          g0 = wmma_bf16(aC[kk], bB0, g0);
          if (two) {
            v16bf bB1 =
                frag_krow(&sB[((jj + 1) * 16 + lane16) * DSTATE + kk * 32], laneHi);
            g1 = wmma_bf16(aC[kk], bB1, g1);
          }
        }
        // weight + causal mask, stash 16x32 bf16 panel in this wave's LDS slot
#pragma unroll
        for (int r = 0; r < 8; r++) {
          int mrow  = r + (laneHi ? 8 : 0);
          int iloc  = mrowBase + mrow;
          int jloc0 = jj * 16 + lane16;
          float w0 = (jloc0 <= iloc)
                         ? g0[r] * __expf(sCum[iloc] - sCum[jloc0]) * sDt[jloc0]
                         : 0.f;
          sMw[mrow * 32 + lane16] = f2bfbits(w0);
          int jloc1 = jloc0 + 16;
          float w1 = (two && jloc1 <= iloc)
                         ? g1[r] * __expf(sCum[iloc] - sCum[jloc1]) * sDt[jloc1]
                         : 0.f;
          sMw[mrow * 32 + 16 + lane16] = f2bfbits(w1);
        }
        // Y += M_panel (16x32) * X_panel (32x64); X^T gives contiguous K=t
        v16bf aM = frag_krow(sMw + lane16 * 32, laneHi);
#pragma unroll
        for (int nt = 0; nt < 4; nt++) {
          v16bf bX = frag_krow(&sXT[(nt * 16 + lane16) * TCH + jj * 16], laneHi);
          acc[nt] = wmma_bf16(aM, bX, acc[nt]);
        }
      }

      // epilogue: + D*x, * silu(z), store
#pragma unroll
      for (int nt = 0; nt < 4; nt++)
#pragma unroll
        for (int r = 0; r < 8; r++) {
          int mrow = r + (laneHi ? 8 : 0);
          int tg = l0 + mrowBase + mrow;
          int pg = h * HDIM + nt * 16 + lane16;
          size_t oi = (size_t)(b * L_SEQ + tg) * DIN + pg;
          float xv = xf[oi];
          float yv = acc[nt][r] + dpar * xv;
          float zv = zx[(size_t)(b * L_SEQ + tg) * DPROJ + pg];
          yv *= zv / (1.f + __expf(-zv));
          out[oi] = yv;
        }
    }
    __syncthreads();   // all reads of sS done

    // ---- phase B: S = exp(cumTot)*S + Xw^T B  (disjoint tiles per wave) ----
    // Tiles wave*4+q share one p-tile per wave: pt = wave/2, ntl = (wave&1)*4+q,
    // so the weighted-X A-fragment is hoisted out of the q loop.
    {
      const float rdec = __expf(cumTot);
      const int pt    = wave >> 1;
      const int nbase = (wave & 1) * 4;
      v8f sacc[4];
#pragma unroll
      for (int q = 0; q < 4; q++)
#pragma unroll
        for (int r = 0; r < 8; r++) {
          int p = pt * 16 + r + (laneHi ? 8 : 0);
          int n = (nbase + q) * 16 + lane16;
          sacc[q][r] = sS[p * DSTATE + n] * rdec;
        }
#pragma unroll
      for (int kk = 0; kk < 8; kk++) {
        int k0 = kk * 32;
        // A row = p, K = t : Xw[t][p] = x * w[t]; contiguous t in sXT
        const unsigned short* xp = &sXT[(pt * 16 + lane16) * TCH + k0];
        v16bf aX;
#pragma unroll
        for (int j = 0; j < 8; j++) {
          int kx = kmap(2 * j, laneHi);
          unsigned v = *(const unsigned*)(xp + kx);
          float x0 = bfbits2f((unsigned short)(v & 0xffffu)) * sW[k0 + kx];
          float x1 = bfbits2f((unsigned short)(v >> 16)) * sW[k0 + kx + 1];
          aX[2 * j]     = bits2bf(f2bfbits(x0));
          aX[2 * j + 1] = bits2bf(f2bfbits(x1));
        }
#pragma unroll
        for (int q = 0; q < 4; q++) {
          // B col = n, K = t : B[t][n]  (strided rows of sB)
          v16bf bB = frag_kstride(&sB[k0 * DSTATE + (nbase + q) * 16 + lane16],
                                  DSTATE, laneHi);
          sacc[q] = wmma_bf16(aX, bB, sacc[q]);
        }
      }
#pragma unroll
      for (int q = 0; q < 4; q++)
#pragma unroll
        for (int r = 0; r < 8; r++) {
          int p = pt * 16 + r + (laneHi ? 8 : 0);
          int n = (nbase + q) * 16 + lane16;
          sS[p * DSTATE + n] = sacc[q][r];
        }
    }
    __syncthreads();   // next chunk overwrites sB/sXT/sCum via TDM + phase 0
  }
}

// ---------------- host launcher ----------------
extern "C" void kernel_launch(void* const* d_in, const int* in_sizes, int n_in,
                              void* d_out, int out_size, void* d_ws, size_t ws_size,
                              hipStream_t stream) {
  const float* zx        = (const float*)d_in[0];
  const float* conv_w    = (const float*)d_in[1];
  const float* conv_b    = (const float*)d_in[2];
  const float* dt_bias   = (const float*)d_in[3];
  const float* a_log     = (const float*)d_in[4];
  const float* d_param   = (const float*)d_in[5];
  const float* dt_scale  = (const float*)d_in[6];
  const float* init_st   = (const float*)d_in[7];
  const int*   dtmin_p   = (const int*)d_in[11];
  const int*   dtmax_p   = (const int*)d_in[12];

  const int Bsz = in_sizes[0] / (L_SEQ * DPROJ);
  const size_t nBL = (size_t)Bsz * L_SEQ;

  auto align256 = [](size_t x) { return (x + 255) & ~(size_t)255; };
  char* w = (char*)d_ws;
  float* xf = (float*)w;            w += align256(nBL * DIN * sizeof(float));
  unsigned short* xbfT = (unsigned short*)w; w += align256(nBL * DIN * 2);
  unsigned short* Bbf = (unsigned short*)w; w += align256(nBL * DSTATE * 2);
  unsigned short* Cbf = (unsigned short*)w; w += align256(nBL * DSTATE * 2);
  float* dtw = (float*)w;           w += align256(nBL * NHEAD * sizeof(float));
  float* dAl = (float*)w;           w += align256(nBL * NHEAD * sizeof(float));

  {
    int total = (int)(nBL * CCH);
    int blocks = (total + 255) / 256;
    conv_silu_kernel<<<blocks, 256, 0, stream>>>(zx, conv_w, conv_b, xf, xbfT,
                                                 Bbf, Cbf, total);
  }
  {
    int total = (int)(nBL * NHEAD);
    int blocks = (total + 255) / 256;
    dt_kernel<<<blocks, 256, 0, stream>>>(zx, dt_bias, a_log, dt_scale,
                                          dtmin_p, dtmax_p, dtw, dAl, total);
  }
  {
    dim3 grid(Bsz * NHEAD);
    ssd_chunk_kernel<<<grid, 256, 0, stream>>>(zx, xf, xbfT, Bbf, Cbf, dtw, dAl,
                                               d_param, init_st, (float*)d_out);
  }
}
